// GHTTokenizer_11553462026733
// MI455X (gfx1250) — matrix operand order repaired and checked by reading, not compiled, not run on hardware
//
#include <hip/hip_runtime.h>

typedef __attribute__((ext_vector_type(16))) _Float16 v16h;
typedef __attribute__((ext_vector_type(8)))  float    v8f;
typedef __attribute__((ext_vector_type(4)))  int      v4i_t;

#define B_   8
#define N_   2048
#define D_   256
#define K_   8
#define NP_  1024
#define F_   512   // 2*D
#define H_   128

#define WMMA_F16(A, Bb, C) \
  __builtin_amdgcn_wmma_f32_16x16x32_f16(false, (A), false, (Bb), (short)0, (C), false, false)

// ---------------- global -> LDS 16B copy, async (CDNA5 async-to-LDS path) if available ----
#if __has_builtin(__builtin_amdgcn_global_load_async_to_lds_b128)
#define HAVE_ASYNC_LDS 1
static __device__ inline void g2l_b128(const void* g, void* l) {
  __builtin_amdgcn_global_load_async_to_lds_b128(
      (__attribute__((address_space(1))) v4i_t*)g,
      (__attribute__((address_space(3))) v4i_t*)l, 0, 0);
}
static __device__ inline void g2l_fence() {
#if __has_builtin(__builtin_amdgcn_s_wait_asynccnt)
  __builtin_amdgcn_s_wait_asynccnt(0);
#else
  asm volatile("s_wait_asynccnt 0" ::: "memory");
#endif
}
#else
#define HAVE_ASYNC_LDS 0
static __device__ inline void g2l_b128(const void* g, void* l) {
  *(uint4*)l = *(const uint4*)g;
}
static __device__ inline void g2l_fence() {}
#endif

// ---------------- fragment load helpers (CDNA5 wave32 WMMA layouts) -------------
// A (16x32 f16): lane m = l&15 ; halves 0..7 -> K = kb..kb+7, halves 8..15 -> K = kb+16..kb+23,
// kb = (l>>4)*8. Two contiguous 8-half runs -> two 16B loads.
static __device__ inline v16h load_a_rows(const _Float16* base, int ld, int lane, int kk) {
  union { v16h v; uint4 q[2]; } f;
  const _Float16* p = base + (size_t)(lane & 15) * ld + kk * 32 + ((lane >> 4) << 3);
  f.q[0] = *reinterpret_cast<const uint4*>(p);
  f.q[1] = *reinterpret_cast<const uint4*>(p + 16);
  return f.v;
}
// B (32x16 f16): lane n = l&15 ; halves 0..7 -> K = kb..kb+7, halves 8..15 -> K = kb+16..kb+23, kb = (l>>4)*8.
static __device__ inline v16h load_b_rows(const _Float16* base, int ld, int lane, int kk) {
  union { v16h v; uint4 q[2]; } f;
  const _Float16* p = base + (size_t)(lane & 15) * ld + kk * 32 + ((lane >> 4) << 3);
  f.q[0] = *reinterpret_cast<const uint4*>(p);
  f.q[1] = *reinterpret_cast<const uint4*>(p + 16);
  return f.v;
}
// B where source is row-major [K x n] (element (k,n) = base[k*ld + n]) -> strided gather.
static __device__ inline v16h load_b_colgather(const _Float16* base, int ld, int lane, int kk) {
  union { v16h v; _Float16 h[16]; } f;
  int n = lane & 15; int kb = kk * 32 + ((lane >> 4) << 3);
#pragma unroll
  for (int t = 0; t < 16; ++t) f.h[t] = base[(size_t)(kb + ((t >> 3) << 4) + (t & 7)) * ld + n];
  return f.v;
}
// A where source is column-major (element (m,k) = base[k*ld + m]).
static __device__ inline v16h load_a_colmajor(const _Float16* base, int ld, int lane, int kk) {
  union { v16h v; _Float16 h[16]; } f;
  int m = lane & 15; int kb = kk * 32 + ((lane >> 4) << 3);
#pragma unroll
  for (int t = 0; t < 16; ++t) {
    int k = kb + ((t >> 3) << 4) + (t & 7);
    f.h[t] = base[(size_t)k * ld + m];
  }
  return f.v;
}

static __device__ inline float wred(float v) {
#pragma unroll
  for (int m = 16; m > 0; m >>= 1) v += __shfl_xor(v, m, 32);
  return v;
}
static __device__ inline float wmaxr(float v) {
#pragma unroll
  for (int m = 16; m > 0; m >>= 1) v = fmaxf(v, __shfl_xor(v, m, 32));
  return v;
}

// ---------------- K0: zero scalar accumulators ----------------
__global__ void k_zero(float* out) {
  if (threadIdx.x == 0) {
    out[(size_t)B_ * NP_ * D_ + (size_t)B_ * NP_ + 0] = 0.0f;  // link_loss
    out[(size_t)B_ * NP_ * D_ + (size_t)B_ * NP_ + 1] = 0.0f;  // entropy
  }
}

// ---------------- K1: normalize tokens -> tn_h (f16); raw tokens -> feat_h[:, :D] ----------------
__global__ void k_prep(const float* __restrict__ tok, _Float16* __restrict__ tn,
                       _Float16* __restrict__ feat) {
  int lane = threadIdx.x & 31;
  int g = blockIdx.x * 8 + (threadIdx.x >> 5);     // token row in [0, B*N)
  const float* p = tok + (size_t)g * D_;
  float v[8]; float ss = 0.0f;
#pragma unroll
  for (int s = 0; s < 8; ++s) { v[s] = p[lane + (s << 5)]; ss += v[s] * v[s]; }
  ss = wred(ss);
  float inv = 1.0f / (sqrtf(ss) + 1e-8f);
#pragma unroll
  for (int s = 0; s < 8; ++s) {
    int d = lane + (s << 5);
    tn[(size_t)g * D_ + d]   = (_Float16)(v[s] * inv);
    feat[(size_t)g * F_ + d] = (_Float16)v[s];
  }
}

// ---------------- K1b: convert W1/W2 to f16 ----------------
__global__ void k_wcvt(const float* __restrict__ W1, const float* __restrict__ W2,
                       _Float16* __restrict__ w1h, _Float16* __restrict__ w2h) {
  int i = blockIdx.x * 256 + threadIdx.x;
  if (i < F_ * H_)  w1h[i] = (_Float16)W1[i];
  if (i < H_ * NP_) w2h[i] = (_Float16)W2[i];
}

// ---------------- K2: kNN top-8 via fused WMMA sim tiles ----------------
__global__ void k_knn(const _Float16* __restrict__ tn, int* __restrict__ knn) {
  __shared__ float stile[4][16][17];
  __shared__ float tv[4][16][8];
  __shared__ int   ti[4][16][8];
  int tid = threadIdx.x, lane = tid & 31, wv = tid >> 5;
  int b = blockIdx.x >> 7, it = blockIdx.x & 127, i0 = it * 16;
  const _Float16* base = tn + ((size_t)b * N_) * D_;

  for (int e = tid; e < 4 * 16 * 8; e += 128) { ((float*)tv)[e] = -3.0e38f; ((int*)ti)[e] = 0; }
  __syncthreads();

  v16h afr[8];
#pragma unroll
  for (int kk = 0; kk < 8; ++kk) afr[kk] = load_a_rows(base + (size_t)i0 * D_, D_, lane, kk);

  for (int jt = wv; jt < 128; jt += 4) {          // equal trip count for all 4 waves
    int j0 = jt * 16;
    // preload ALL 8 B fragments first: one long load clause, staggered waits,
    // then 8 back-to-back WMMAs (latency hidden behind the matrix pipe)
    v16h bfr[8];
#pragma unroll
    for (int kk = 0; kk < 8; ++kk) bfr[kk] = load_b_rows(base + (size_t)j0 * D_, D_, lane, kk);
    v8f c = {};
#pragma unroll
    for (int kk = 0; kk < 8; ++kk) c = WMMA_F16(afr[kk], bfr[kk], c);
#pragma unroll
    for (int r = 0; r < 8; ++r) {
      int m = r + ((lane >> 4) << 3);
      stile[wv][m][lane & 15] = c[r];
    }
    __syncthreads();
    if (lane < 16) {
      int m = lane, ri = i0 + m;
      for (int n = 0; n < 16; ++n) {
        int col = j0 + n;
        float v = (col == ri) ? -3.0e38f : stile[wv][m][n];
        if (v > tv[wv][m][7]) {                    // insert into sorted-desc list of 8
          int e = 7;
          while (e > 0 && tv[wv][m][e - 1] < v) {
            tv[wv][m][e] = tv[wv][m][e - 1];
            ti[wv][m][e] = ti[wv][m][e - 1];
            --e;
          }
          tv[wv][m][e] = v; ti[wv][m][e] = col;
        }
      }
    }
    __syncthreads();
  }
  __syncthreads();
  if (tid < 16) {                                  // merge the 4 per-wave lists
    int m = tid;
    for (int e = 0; e < 8; ++e) {
      float best = -3.0e38f; int bw = 0, be = 0;
      for (int w2 = 0; w2 < 4; ++w2)
        for (int q = 0; q < 8; ++q)
          if (tv[w2][m][q] > best) { best = tv[w2][m][q]; bw = w2; be = q; }
      knn[((size_t)b * N_ + i0 + m) * K_ + e] = ti[bw][m][be];
      tv[bw][m][be] = -3.0e38f;
    }
  }
}

// ---------------- K3: mutual check + degree-normalized aggregation -> feat_h[:, D:2D] ----------------
__global__ void k_agg(const float* __restrict__ tok, const int* __restrict__ knn,
                      _Float16* __restrict__ feat) {
  int lane = threadIdx.x & 31;
  int g = blockIdx.x * 8 + (threadIdx.x >> 5);
  int b = g >> 11, i = g & (N_ - 1);
  const int* kp = knn + (size_t)g * K_;
  int kj = kp[lane & 7];
  float acc[8] = {0, 0, 0, 0, 0, 0, 0, 0};
  int deg = 0;
  for (int jj = 0; jj < 8; ++jj) {
    int j = __shfl(kj, jj, 32);
    int pred = 0;
    if (lane < 8) pred = (knn[((size_t)b * N_ + j) * K_ + lane] == i);
    if (__any(pred)) {
      ++deg;
      const float* tp = tok + ((size_t)b * N_ + j) * D_;
#pragma unroll
      for (int s = 0; s < 8; ++s) acc[s] += tp[lane + (s << 5)];
    }
  }
  float inv = 1.0f / (float)(deg < 1 ? 1 : deg);
#pragma unroll
  for (int s = 0; s < 8; ++s)
    feat[(size_t)g * F_ + D_ + lane + (s << 5)] = (_Float16)(acc[s] * inv);
}

// ---------------- K4: H = gelu(feat @ W1 + b1) ----------------
__global__ void k_mlp1(const _Float16* __restrict__ feat, const _Float16* __restrict__ w1,
                       const float* __restrict__ b1, _Float16* __restrict__ hh) {
  int lane = threadIdx.x & 31, wv = threadIdx.x >> 5;
  int row0 = blockIdx.x * 16;
  const _Float16* abase = feat + (size_t)row0 * F_;
  v8f c = {};
  for (int kk = 0; kk < 16; ++kk) {
    v16h a  = load_a_rows(abase, F_, lane, kk);
    v16h bb = load_b_colgather(w1 + wv * 16, H_, lane, kk);
    c = WMMA_F16(a, bb, c);
  }
#pragma unroll
  for (int r = 0; r < 8; ++r) {
    int m = r + ((lane >> 4) << 3);
    int col = wv * 16 + (lane & 15);
    float x = c[r] + b1[col];
    float g = 0.5f * x * (1.0f + erff(x * 0.70710678118f));   // exact GELU
    hh[(size_t)(row0 + m) * H_ + col] = (_Float16)g;
  }
}

// ---------------- K5: logits = H @ W2 + b2, softmax, entropy ----------------
__global__ void k_mlp2(const _Float16* __restrict__ hh, const _Float16* __restrict__ w2,
                       const float* __restrict__ b2, _Float16* __restrict__ assign,
                       float* __restrict__ ent_out) {
  __shared__ float slog[16][NP_];                   // 64 KB
  int lane = threadIdx.x & 31, wv = threadIdx.x >> 5;
  int row0 = blockIdx.x * 16;
  const _Float16* abase = hh + (size_t)row0 * H_;
  v16h afr[4];
#pragma unroll
  for (int kk = 0; kk < 4; ++kk) afr[kk] = load_a_rows(abase, H_, lane, kk);

  for (int ct = 0; ct < 8; ++ct) {
    int col0 = wv * 128 + ct * 16;
    v8f c = {};
#pragma unroll
    for (int kk = 0; kk < 4; ++kk) {
      v16h bb = load_b_colgather(w2 + col0, NP_, lane, kk);
      c = WMMA_F16(afr[kk], bb, c);
    }
#pragma unroll
    for (int r = 0; r < 8; ++r) {
      int m = r + ((lane >> 4) << 3);
      int col = col0 + (lane & 15);
      slog[m][col] = c[r] + b2[col];
    }
  }
  __syncthreads();
  for (int rr = 0; rr < 2; ++rr) {                  // 8 waves x 2 rows = 16 rows
    int m = wv * 2 + rr;
    size_t row = (size_t)row0 + m;
    float mx = -3.0e38f;
    for (int cc = lane; cc < NP_; cc += 32) mx = fmaxf(mx, slog[m][cc]);
    mx = wmaxr(mx);
    float s = 0.0f;
    for (int cc = lane; cc < NP_; cc += 32) s += __expf(slog[m][cc] - mx);
    s = wred(s);
    float inv = 1.0f / s, ent = 0.0f;
    for (int cc = lane; cc < NP_; cc += 32) {
      float p = __expf(slog[m][cc] - mx) * inv;
      assign[row * NP_ + cc] = (_Float16)p;
      ent += -p * __logf(p + 1e-8f);
    }
    ent = wred(ent);
    if (lane == 0) atomicAdd(ent_out, ent * (1.0f / ((float)B_ * N_)));
  }
}

// ---------------- K6: pooled = assign^T @ tokens ----------------
// All 8 waves of a block share (b, p0): stage assign[:, p0:p0+16] (64 KB) in LDS with
// coalesced/async b128 loads; A fragments then come from LDS (ds_load), B from global.
__global__ void k_pool(const _Float16* __restrict__ assign, const _Float16* __restrict__ feat,
                       float* __restrict__ pooled) {
  __shared__ _Float16 astage[N_ * 16];              // 64 KB, layout [n][16]
  int tid = threadIdx.x, lane = tid & 31, wv = tid >> 5;
  int jb = blockIdx.x * 8;
  int b  = jb >> 10;
  int p0 = (((jb & 1023) >> 4)) * 16;
  int d0 = ((jb & 15) + wv) * 16;

  const _Float16* srcp = assign + ((size_t)b * N_) * NP_ + p0;
  for (int t = tid; t < N_ * 2; t += 256)           // 2 x b128 per n-row
    g2l_b128(srcp + (size_t)(t >> 1) * NP_ + (t & 1) * 8, &astage[(t >> 1) * 16 + (t & 1) * 8]);
  g2l_fence();
  __syncthreads();

  const _Float16* bbase = feat + ((size_t)b * N_) * F_ + d0;   // B(k,n)=tok[k][d0+n]
  v8f c = {};
  for (int kk = 0; kk < 64; ++kk) {
    v16h a  = load_a_colmajor(astage, 16, lane, kk);           // ds gathers
    v16h bb = load_b_colgather(bbase, F_, lane, kk);
    c = WMMA_F16(a, bb, c);
  }
#pragma unroll
  for (int r = 0; r < 8; ++r) {
    int m = r + ((lane >> 4) << 3);
    pooled[((size_t)b * NP_ + p0 + m) * D_ + d0 + (lane & 15)] = c[r];
  }
}

// ---------------- K6b: pooled_t = assign^T @ timestamps ----------------
__global__ void k_poolt(const _Float16* __restrict__ assign, const float* __restrict__ ts,
                        float* __restrict__ pt) {
  int lane = threadIdx.x & 31;
  int idx = blockIdx.x * 8 + (threadIdx.x >> 5);    // 8192 waves
  int b = idx >> 10, p = idx & 1023;
  float acc = 0.0f;
  for (int n = lane; n < N_; n += 32)
    acc += (float)assign[((size_t)b * N_ + n) * NP_ + p] * ts[(size_t)b * N_ + n];
  acc = wred(acc);
  if (lane == 0) pt[(size_t)b * NP_ + p] = acc;
}

// ---------------- K7: bitonic argsort by pooled_t + gather into d_out ----------------
__global__ void k_sort(const float* __restrict__ pooled, const float* __restrict__ pt,
                       float* __restrict__ out) {
  __shared__ float skey[NP_];
  __shared__ int   sidx[NP_];
  int tid = threadIdx.x;                            // 512 threads
  int b = blockIdx.x;
  for (int i = tid; i < NP_; i += 512) { skey[i] = pt[(size_t)b * NP_ + i]; sidx[i] = i; }
  __syncthreads();
  for (int ks = 2; ks <= NP_; ks <<= 1) {
    for (int j = ks >> 1; j > 0; j >>= 1) {
      for (int i = tid; i < NP_; i += 512) {
        int ixj = i ^ j;
        if (ixj > i) {
          bool up = ((i & ks) == 0);
          float a = skey[i], c = skey[ixj];
          if ((a > c) == up) {
            skey[i] = c; skey[ixj] = a;
            int t = sidx[i]; sidx[i] = sidx[ixj]; sidx[ixj] = t;
          }
        }
      }
      __syncthreads();
    }
  }
  float* outTok = out;
  float* outT   = out + (size_t)B_ * NP_ * D_;
  for (int p = tid; p < NP_; p += 512) outT[(size_t)b * NP_ + p] = skey[p];
  for (int e = tid; e < NP_ * D_; e += 512) {
    int p = e >> 8, d = e & 255;
    outTok[((size_t)b * NP_ + p) * D_ + d] = pooled[((size_t)b * NP_ + sidx[p]) * D_ + d];
  }
}

// ---------------- K8: link_loss = mean((assign @ assign^T - adj)^2) ----------------
__global__ void k_link(const _Float16* __restrict__ assign, const int* __restrict__ knn,
                       float* __restrict__ link_out) {
  __shared__ _Float16 astage[16][NP_];              // 32 KB: 16 assign rows, reused 128x
  __shared__ int knni[16][8];
  int tid = threadIdx.x, lane = tid & 31, wv = tid >> 5;
  int b = blockIdx.x >> 7, it = blockIdx.x & 127, i0 = it * 16;

  const _Float16* srcp = assign + ((size_t)b * N_ + i0) * NP_;   // contiguous 32 KB
  for (int t = tid; t < 16 * (NP_ / 8); t += 256)                // 2048 x b128
    g2l_b128(srcp + (size_t)t * 8, &astage[0][0] + (size_t)t * 8);
  g2l_fence();
  for (int e = tid; e < 16 * 8; e += 256)
    knni[e >> 3][e & 7] = knn[((size_t)b * N_ + i0 + (e >> 3)) * K_ + (e & 7)];
  __syncthreads();

  float acc = 0.0f;
  for (int jt = wv; jt < 128; jt += 8) {
    int j0 = jt * 16;
    int j = j0 + (lane & 15);
    int knnj[8];
    const int* kpj = knn + ((size_t)b * N_ + j) * K_;
#pragma unroll
    for (int e = 0; e < 8; ++e) knnj[e] = kpj[e];

    const _Float16* bbase = assign + ((size_t)b * N_ + j0) * NP_;
    __builtin_prefetch(bbase + (size_t)16 * NP_, 0, 1);
    v8f c = {};
    v16h bcur = load_b_rows(bbase, NP_, lane, 0);   // software pipeline: prefetch next B
    for (int kk = 0; kk < 32; ++kk) {
      v16h bnext = load_b_rows(bbase, NP_, lane, (kk + 1) & 31);
      v16h a = load_a_rows(&astage[0][0], NP_, lane, kk);        // ds_load from LDS
      c = WMMA_F16(a, bcur, c);
      bcur = bnext;
    }
#pragma unroll
    for (int r = 0; r < 8; ++r) {
      int m = r + ((lane >> 4) << 3);
      int i = i0 + m;
      bool a1 = false, a2 = false;
#pragma unroll
      for (int e = 0; e < 8; ++e) { a1 |= (knni[m][e] == j); a2 |= (knnj[e] == i); }
      float dlt = c[r] - ((a1 && a2) ? 1.0f : 0.0f);
      acc += dlt * dlt;
    }
  }
  acc = wred(acc);
  if (lane == 0) atomicAdd(link_out, acc * (1.0f / ((float)B_ * (float)N_ * (float)N_)));
}

// ---------------- host ----------------
extern "C" void kernel_launch(void* const* d_in, const int* in_sizes, int n_in,
                              void* d_out, int out_size, void* d_ws, size_t ws_size,
                              hipStream_t stream) {
  const float* tokens = (const float*)d_in[0];
  const float* ts     = (const float*)d_in[1];
  const float* W1     = (const float*)d_in[2];
  const float* b1     = (const float*)d_in[3];
  const float* W2     = (const float*)d_in[4];
  const float* b2     = (const float*)d_in[5];
  float* out = (float*)d_out;

  char* w = (char*)d_ws;
  auto alloc = [&](size_t bytes) -> void* {
    void* p = (void*)w;
    w += (bytes + 255) & ~(size_t)255;
    return p;
  };
  _Float16* tn_h     = (_Float16*)alloc((size_t)B_ * N_ * D_ * 2);   // 8 MB
  _Float16* feat_h   = (_Float16*)alloc((size_t)B_ * N_ * F_ * 2);   // 16 MB
  _Float16* h_h      = (_Float16*)alloc((size_t)B_ * N_ * H_ * 2);   // 4 MB
  _Float16* assign_h = (_Float16*)alloc((size_t)B_ * N_ * NP_ * 2);  // 32 MB
  int*      knn      = (int*)alloc((size_t)B_ * N_ * K_ * 4);        // 512 KB
  float*    pooled_w = (float*)alloc((size_t)B_ * NP_ * D_ * 4);     // 8 MB
  float*    pt_w     = (float*)alloc((size_t)B_ * NP_ * 4);          // 32 KB
  _Float16* w1_h     = (_Float16*)alloc((size_t)F_ * H_ * 2);
  _Float16* w2_h     = (_Float16*)alloc((size_t)H_ * NP_ * 2);

  float* link_out = out + (size_t)B_ * NP_ * D_ + (size_t)B_ * NP_;
  float* ent_out  = link_out + 1;

  k_zero <<<1, 32, 0, stream>>>(out);
  k_prep <<<(B_ * N_) / 8, 256, 0, stream>>>(tokens, tn_h, feat_h);
  k_wcvt <<<512, 256, 0, stream>>>(W1, W2, w1_h, w2_h);
  k_knn  <<<B_ * (N_ / 16), 128, 0, stream>>>(tn_h, knn);
  k_agg  <<<(B_ * N_) / 8, 256, 0, stream>>>(tokens, knn, feat_h);
  k_mlp1 <<<(B_ * N_) / 16, 256, 0, stream>>>(feat_h, w1_h, b1, h_h);
  k_mlp2 <<<(B_ * N_) / 16, 256, 0, stream>>>(h_h, w2_h, b2, assign_h, ent_out);
  k_pool <<<1024, 256, 0, stream>>>(assign_h, feat_h, pooled_w);
  k_poolt<<<1024, 256, 0, stream>>>(assign_h, ts, pt_w);
  k_sort <<<B_, 512, 0, stream>>>(pooled_w, pt_w, out);
  k_link <<<B_ * (N_ / 16), 256, 0, stream>>>(assign_h, knn, link_out);

  (void)in_sizes; (void)n_in; (void)out_size; (void)ws_size;
}
